// Model_19052474925724
// MI455X (gfx1250) — compile-verified, matrix operand order checked by reading, not run on hardware
//
#include <hip/hip_runtime.h>

// ---------------------------------------------------------------------------
// Bi-Mamba time-series forecaster for MI455X (gfx1250, wave32, WMMA).
// GEMMs: v_wmma_f32_16x16x32_f16, 64x128 block tile, 32x32 wave tile,
// double-buffered LDS, GLOBAL_LOAD_ASYNC_TO_LDS_B128 staging (ASYNCcnt).
// ---------------------------------------------------------------------------

typedef __attribute__((ext_vector_type(16))) _Float16 v16h;
typedef __attribute__((ext_vector_type(8)))  _Float16 v8h;
typedef __attribute__((ext_vector_type(8)))  float    v8f;
typedef __attribute__((ext_vector_type(4)))  int      v4i;

static constexpr int kB    = 32;
static constexpr int kSeq  = 512;
static constexpr int kNvar = 256;
static constexpr int kMark = 6;
static constexpr int kTok  = kNvar + kMark;   // 262
static constexpr int kD    = 512;
static constexpr int kS    = 16;
static constexpr int kR    = 32;
static constexpr int kPred = 96;
static constexpr int kTT   = kB * kTok;       // 8384 GEMM rows

// ---- async global->LDS staging (builtin exists on this toolchain) ----------
#ifndef USE_ASYNC_LDS
#define USE_ASYNC_LDS 1
#endif
#if USE_ASYNC_LDS && __has_builtin(__builtin_amdgcn_global_load_async_to_lds_b128)
#define ASYNC_LDS 1
#else
#define ASYNC_LDS 0
#endif

#if ASYNC_LDS
typedef __attribute__((address_space(1))) v4i* gptr_v4i;
typedef __attribute__((address_space(3))) v4i* lptr_v4i;

__device__ __forceinline__ void async_copy16(const _Float16* g, _Float16* l)
{
  // global pointer: full 64-bit address; LDS pointer: low 32 bits of the
  // generic address are the LDS byte offset (flat->LDS truncates addr[31:0]).
  gptr_v4i gp = (gptr_v4i)(unsigned long long)(uintptr_t)g;
  lptr_v4i lp = (lptr_v4i)(unsigned int)(uintptr_t)l;
  __builtin_amdgcn_global_load_async_to_lds_b128(gp, lp, 0, 0);
}
__device__ __forceinline__ void wait_async0()
{
#if __has_builtin(__builtin_amdgcn_s_wait_asynccnt)
  __builtin_amdgcn_s_wait_asynccnt(0);
#else
  asm volatile("s_wait_asynccnt 0" ::: "memory");
#endif
}
#endif

// ---------------------------------------------------------------------------
// f32 -> f16 cast (weights, once per launch)
// ---------------------------------------------------------------------------
__global__ void cast_f16_kernel(const float* __restrict__ in,
                                _Float16* __restrict__ out, int n)
{
  for (int i = blockIdx.x * blockDim.x + threadIdx.x; i < n;
       i += gridDim.x * blockDim.x)
    out[i] = (_Float16)in[i];
}

// ---------------------------------------------------------------------------
// WMMA GEMM:  C[M,N] = act( A[M,K] @ W[N,K]^T + bias (+ C) )
//   A, W f16 row-major; C (f32) / C16 (f16) each optional.
//   act: 0 none, 1 relu, 2 softplus.  addC: accumulate into C.
// Block 256 thr = 8 waves; tile 64(M) x 128(N); wave tile 32x32; TK=32;
// double-buffered LDS; one barrier per K-step.
// Preconditions: K % 32 == 0; lda/ldw rows 16-B aligned (all call sites ok).
// ---------------------------------------------------------------------------
__device__ __forceinline__ void gemm_emit(float v, int row, int col,
    const float* __restrict__ bias, float* __restrict__ C, int ldc,
    _Float16* __restrict__ C16, int ldc16, int act, int addC)
{
  if (bias) v += bias[col];
  if (addC && C) v += C[(size_t)row * ldc + col];
  if (act == 1)      v = v > 0.f ? v : 0.f;
  else if (act == 2) v = (v > 20.f) ? v : log1pf(__expf(v));
  if (C)   C[(size_t)row * ldc + col] = v;
  if (C16) C16[(size_t)row * ldc16 + col] = (_Float16)v;
}

// ISA 16-bit fragment: lane = row (within 16-row group), half-wave selects
// K 0..7/8..15 (lo) and 16..23/24..31 (hi) -> two contiguous 16-B reads.
__device__ __forceinline__ v16h load_frag(const _Float16* p, int half)
{
  v8h lo = *(const v8h*)(p + half * 8);
  v8h hi = *(const v8h*)(p + 16 + half * 8);
  v16h f;
  #pragma unroll
  for (int i = 0; i < 8; ++i) { f[i] = lo[i]; f[i + 8] = hi[i]; }
  return f;
}

__global__ __launch_bounds__(256) void gemm_wmma_kernel(
    const _Float16* __restrict__ A, int lda,
    const _Float16* __restrict__ W, int ldw,
    const float* __restrict__ bias,
    float* __restrict__ C, int ldc,
    _Float16* __restrict__ C16, int ldc16,
    int M, int N, int K, int act, int addC)
{
  constexpr int BM = 64, BN = 128, TK = 32;
  constexpr int LDSA = 40;                 // 80-B rows: 16-B aligned, 0-conflict
  __shared__ _Float16 As[2][BM * LDSA];
  __shared__ _Float16 Ws[2][BN * LDSA];

  const int tid  = threadIdx.x;
  const int lane = tid & 31;
  const int wave = tid >> 5;
  const int wm   = wave >> 2;              // 0..1
  const int wn   = wave & 3;               // 0..3
  const int half = lane >> 4;
  const int r    = lane & 15;
  const int tileM = blockIdx.y * BM;
  const int tileN = blockIdx.x * BN;

  // staging: A tile 64x32 -> 1 chunk/thread; W tile 128x32 -> 2 chunks/thread
  const int srow = tid >> 2;               // 0..63
  const int scol = (tid & 3) << 3;         // 0,8,16,24
  // clamp OOB rows to a valid row: garbage feeds only discarded columns
  const int agr  = min(tileM + srow, M - 1);
  const int wgr0 = min(tileN + srow, N - 1);
  const int wgr1 = min(tileN + 64 + srow, N - 1);
  const _Float16* gA  = A + (size_t)agr  * lda + scol;
  const _Float16* gW0 = W + (size_t)wgr0 * ldw + scol;
  const _Float16* gW1 = W + (size_t)wgr1 * ldw + scol;
  const int lA  = srow * LDSA + scol;
  const int lW0 = srow * LDSA + scol;
  const int lW1 = (64 + srow) * LDSA + scol;

  v8f acc00 = {}, acc01 = {}, acc10 = {}, acc11 = {};

  // ---- prologue: stage k=0 into buffer 0 ----
#if ASYNC_LDS
  async_copy16(gA,  &As[0][lA]);
  async_copy16(gW0, &Ws[0][lW0]);
  async_copy16(gW1, &Ws[0][lW1]);
  wait_async0();
#else
  *(v8h*)&As[0][lA]  = *(const v8h*)gA;
  *(v8h*)&Ws[0][lW0] = *(const v8h*)gW0;
  *(v8h*)&Ws[0][lW1] = *(const v8h*)gW1;
#endif
  __syncthreads();

  int cur = 0;
  for (int k0 = 0; k0 < K; k0 += TK) {
    const int nxt = k0 + TK;
    const int nb  = cur ^ 1;

    // ---- kick off next tile while this one computes ----
#if ASYNC_LDS
    if (nxt < K) {
      async_copy16(gA + nxt,  &As[nb][lA]);
      async_copy16(gW0 + nxt, &Ws[nb][lW0]);
      async_copy16(gW1 + nxt, &Ws[nb][lW1]);
    }
#else
    v8h ra = {}, rw0 = {}, rw1 = {};
    if (nxt < K) {
      ra  = *(const v8h*)(gA + nxt);
      rw0 = *(const v8h*)(gW0 + nxt);
      rw1 = *(const v8h*)(gW1 + nxt);
    }
#endif

    // ---- 4 WMMAs on current buffer ----
    const v16h a0 = load_frag(&As[cur][(wm * 32 + r) * LDSA], half);
    const v16h a1 = load_frag(&As[cur][(wm * 32 + 16 + r) * LDSA], half);
    const v16h b0 = load_frag(&Ws[cur][(wn * 32 + r) * LDSA], half);
    const v16h b1 = load_frag(&Ws[cur][(wn * 32 + 16 + r) * LDSA], half);
    acc00 = __builtin_amdgcn_wmma_f32_16x16x32_f16(false, a0, false, b0,
                                                   (short)0, acc00, false, false);
    acc01 = __builtin_amdgcn_wmma_f32_16x16x32_f16(false, a0, false, b1,
                                                   (short)0, acc01, false, false);
    acc10 = __builtin_amdgcn_wmma_f32_16x16x32_f16(false, a1, false, b0,
                                                   (short)0, acc10, false, false);
    acc11 = __builtin_amdgcn_wmma_f32_16x16x32_f16(false, a1, false, b1,
                                                   (short)0, acc11, false, false);

#if ASYNC_LDS
    if (nxt < K) wait_async0();
#else
    if (nxt < K) {
      *(v8h*)&As[nb][lA]  = ra;
      *(v8h*)&Ws[nb][lW0] = rw0;
      *(v8h*)&Ws[nb][lW1] = rw1;
    }
#endif
    __syncthreads();
    cur = nb;
  }

  // ---- epilogue (C layout: vgpr i -> row half*8+i, lane&15 -> col) ----
  const int row0 = tileM + wm * 32 + half * 8;
  const int row1 = row0 + 16;
  const int col0 = tileN + wn * 32 + r;
  const int col1 = col0 + 16;
  #pragma unroll
  for (int i = 0; i < 8; ++i) {
    const int r0 = row0 + i, r1 = row1 + i;
    if (r0 < M) {
      if (col0 < N) gemm_emit(acc00[i], r0, col0, bias, C, ldc, C16, ldc16, act, addC);
      if (col1 < N) gemm_emit(acc01[i], r0, col1, bias, C, ldc, C16, ldc16, act, addC);
    }
    if (r1 < M) {
      if (col0 < N) gemm_emit(acc10[i], r1, col0, bias, C, ldc, C16, ldc16, act, addC);
      if (col1 < N) gemm_emit(acc11[i], r1, col1, bias, C, ldc, C16, ldc16, act, addC);
    }
  }
}

// ---------------------------------------------------------------------------
// RevIN + token build
// ---------------------------------------------------------------------------
__global__ __launch_bounds__(256) void norm_tokens_kernel(
    const float* __restrict__ x_enc, const float* __restrict__ x_mark,
    _Float16* __restrict__ tok16, float* __restrict__ meanb,
    float* __restrict__ stdb)
{
  __shared__ float rs[256];
  __shared__ float rq[256];
  const int blk = blockIdx.x;          // b*kTok + v
  const int b = blk / kTok;
  const int v = blk % kTok;
  const int tid = threadIdx.x;
  _Float16* trow = tok16 + (size_t)blk * kSeq;

  if (v < kNvar) {
    const float x0 = x_enc[((size_t)b * kSeq + tid)       * kNvar + v];
    const float x1 = x_enc[((size_t)b * kSeq + tid + 256) * kNvar + v];
    rs[tid] = x0 + x1;
    rq[tid] = x0 * x0 + x1 * x1;
    __syncthreads();
    for (int s = 128; s > 0; s >>= 1) {
      if (tid < s) { rs[tid] += rs[tid + s]; rq[tid] += rq[tid + s]; }
      __syncthreads();
    }
    const float mu  = rs[0] * (1.f / kSeq);
    const float var = rq[0] * (1.f / kSeq) - mu * mu;
    const float sd  = sqrtf(var + 1e-5f);
    const float inv = 1.f / sd;
    trow[tid]       = (_Float16)((x0 - mu) * inv);
    trow[tid + 256] = (_Float16)((x1 - mu) * inv);
    if (tid == 0) { meanb[blk] = mu; stdb[blk] = sd; }
  } else {
    const int m = v - kNvar;
    trow[tid]       = (_Float16)x_mark[((size_t)b * kSeq + tid)       * kMark + m];
    trow[tid + 256] = (_Float16)x_mark[((size_t)b * kSeq + tid + 256) * kMark + m];
    if (tid == 0) { meanb[blk] = 0.f; stdb[blk] = 1.f; }
  }
}

// ---------------------------------------------------------------------------
// Depth-2 causal conv + SiLU (direction-aware neighbor)
// ---------------------------------------------------------------------------
__global__ void conv_silu_kernel(const float* __restrict__ xz,
    const float* __restrict__ cw, const float* __restrict__ cb,
    float* __restrict__ xc32, _Float16* __restrict__ xc16, int dir)
{
  const int n = kTT * kD;
  for (int i = blockIdx.x * blockDim.x + threadIdx.x; i < n;
       i += gridDim.x * blockDim.x) {
    const int bt = i / kD;
    const int d  = i - bt * kD;
    const int t  = bt % kTok;
    const float cur = xz[(size_t)bt * (2 * kD) + d];
    float prev = 0.f;
    if (dir == 0) { if (t > 0)        prev = xz[(size_t)(bt - 1) * (2 * kD) + d]; }
    else          { if (t < kTok - 1) prev = xz[(size_t)(bt + 1) * (2 * kD) + d]; }
    const float val = prev * cw[d * 2 + 0] + cur * cw[d * 2 + 1] + cb[d];
    const float s = val / (1.f + __expf(-val));
    xc32[i] = s;
    xc16[i] = (_Float16)s;
  }
}

// ---------------------------------------------------------------------------
// Selective scan + gating. One block/batch; lane = channel d; 16-state
// recurrence in registers; 8 timesteps of B/C rows staged per barrier.
// ---------------------------------------------------------------------------
__global__ __launch_bounds__(512) void mamba_scan_kernel(
    const float* __restrict__ dt, const float* __restrict__ xc,
    const float* __restrict__ xz, const float* __restrict__ dbl,
    const float* __restrict__ alog, const float* __restrict__ dparam,
    _Float16* __restrict__ y16, int dir)
{
  constexpr int TST = 8;
  __shared__ float sBC[TST][32];       // [step][0:16)=B, [16:32)=C
  const int b = blockIdx.x;
  const int d = threadIdx.x;

  float hS[kS];
  float Ad[kS];
  #pragma unroll
  for (int s = 0; s < kS; ++s) {
    hS[s] = 0.f;
    Ad[s] = -__expf(alog[d * kS + s]); // A = -exp(alog)
  }
  const float dp = dparam[d];

  for (int t0 = 0; t0 < kTok; t0 += TST) {
    const int nstep = min(TST, kTok - t0);
    __syncthreads();
    if (d < nstep * 32) {
      const int step = d >> 5, idx = d & 31;
      const int t = dir ? (kTok - 1 - (t0 + step)) : (t0 + step);
      sBC[step][idx] = dbl[((size_t)b * kTok + t) * (kR + 2 * kS) + kR + idx];
    }
    __syncthreads();
    for (int ss = 0; ss < nstep; ++ss) {
      const int t = dir ? (kTok - 1 - (t0 + ss)) : (t0 + ss);
      const size_t bt = (size_t)b * kTok + t;
      const float dtv = dt[bt * kD + d];
      const float xcv = xc[bt * kD + d];
      const float zv  = xz[bt * (2 * kD) + kD + d];
      float acc = 0.f;
      #pragma unroll
      for (int s = 0; s < kS; ++s) {
        const float dA = __expf(dtv * Ad[s]);
        hS[s] = dA * hS[s] + dtv * sBC[ss][s] * xcv;
        acc += hS[s] * sBC[ss][kS + s];
      }
      const float yv = acc + dp * xcv;
      const float sz = zv / (1.f + __expf(-zv));   // silu(z)
      y16[bt * kD + d] = (_Float16)(yv * sz);
    }
  }
}

// ---------------------------------------------------------------------------
// LayerNorm over D=512; optional f32 and f16 outputs.
// ---------------------------------------------------------------------------
__global__ __launch_bounds__(256) void layernorm_kernel(
    const float* __restrict__ in, const float* __restrict__ g,
    const float* __restrict__ bb, float* __restrict__ out32,
    _Float16* __restrict__ out16)
{
  __shared__ float rs[256];
  __shared__ float rq[256];
  const int row = blockIdx.x;
  const int tid = threadIdx.x;
  const float x0 = in[(size_t)row * kD + tid];
  const float x1 = in[(size_t)row * kD + tid + 256];
  rs[tid] = x0 + x1;
  rq[tid] = x0 * x0 + x1 * x1;
  __syncthreads();
  for (int s = 128; s > 0; s >>= 1) {
    if (tid < s) { rs[tid] += rs[tid + s]; rq[tid] += rq[tid + s]; }
    __syncthreads();
  }
  const float mu  = rs[0] * (1.f / kD);
  const float var = rq[0] * (1.f / kD) - mu * mu;
  const float inv = rsqrtf(var + 1e-5f);
  const float y0 = (x0 - mu) * inv * g[tid]       + bb[tid];
  const float y1 = (x1 - mu) * inv * g[tid + 256] + bb[tid + 256];
  if (out32) { out32[(size_t)row * kD + tid] = y0;
               out32[(size_t)row * kD + tid + 256] = y1; }
  if (out16) { out16[(size_t)row * kD + tid] = (_Float16)y0;
               out16[(size_t)row * kD + tid + 256] = (_Float16)y1; }
}

// ---------------------------------------------------------------------------
// Transpose projection output + RevIN denorm
// ---------------------------------------------------------------------------
__global__ void final_denorm_kernel(const float* __restrict__ dec,
    const float* __restrict__ meanb, const float* __restrict__ stdb,
    float* __restrict__ out)
{
  const int n = kB * kPred * kNvar;
  for (int i = blockIdx.x * blockDim.x + threadIdx.x; i < n;
       i += gridDim.x * blockDim.x) {
    const int b   = i / (kPred * kNvar);
    const int rem = i - b * (kPred * kNvar);
    const int p   = rem / kNvar;
    const int v   = rem - p * kNvar;
    const int tokidx = b * kTok + v;
    out[i] = dec[(size_t)tokidx * kPred + p] * stdb[tokidx] + meanb[tokidx];
  }
}

// ---------------------------------------------------------------------------
// Host orchestration
// ---------------------------------------------------------------------------
static inline void launch_gemm(hipStream_t stream,
    const _Float16* A, int lda, const _Float16* W, int ldw, const float* bias,
    float* C, int ldc, _Float16* C16, int ldc16,
    int M, int N, int K, int act, int addC)
{
  dim3 grid((N + 127) / 128, (M + 63) / 64);
  gemm_wmma_kernel<<<grid, 256, 0, stream>>>(A, lda, W, ldw, bias,
                                             C, ldc, C16, ldc16,
                                             M, N, K, act, addC);
}

extern "C" void kernel_launch(void* const* d_in, const int* in_sizes, int n_in,
                              void* d_out, int out_size, void* d_ws,
                              size_t ws_size, hipStream_t stream)
{
  (void)in_sizes; (void)n_in; (void)out_size; (void)ws_size;

  const float* x_enc    = (const float*)d_in[0];
  const float* x_mark   = (const float*)d_in[1];
  const float* emb_w    = (const float*)d_in[2];
  const float* emb_b    = (const float*)d_in[3];
  const float* m_win    = (const float*)d_in[4];
  const float* m_convw  = (const float*)d_in[5];
  const float* m_convb  = (const float*)d_in[6];
  const float* m_wx     = (const float*)d_in[7];
  const float* m_wdt    = (const float*)d_in[8];
  const float* m_bdt    = (const float*)d_in[9];
  const float* m_alog   = (const float*)d_in[10];
  const float* m_dparam = (const float*)d_in[11];
  const float* m_wout   = (const float*)d_in[12];
  const float* ln1_g    = (const float*)d_in[13];
  const float* ln1_b    = (const float*)d_in[14];
  const float* ff1_w    = (const float*)d_in[15];
  const float* ff1_b    = (const float*)d_in[16];
  const float* ff2_w    = (const float*)d_in[17];
  const float* ff2_b    = (const float*)d_in[18];
  const float* ln2_g    = (const float*)d_in[19];
  const float* ln2_b    = (const float*)d_in[20];
  const float* lnf_g    = (const float*)d_in[21];
  const float* lnf_b    = (const float*)d_in[22];
  const float* proj_w   = (const float*)d_in[23];
  const float* proj_b   = (const float*)d_in[24];
  float* out = (float*)d_out;

  // ---- workspace carve-out ----
  char* base = (char*)d_ws;
  size_t off = 0;
  auto alloc = [&](size_t bytes) -> void* {
    void* p = base + off;
    off += (bytes + 255) & ~(size_t)255;
    return p;
  };
  const size_t A32 = (size_t)kTT * kD * sizeof(float);
  const size_t A16 = (size_t)kTT * kD * sizeof(_Float16);

  _Float16* tok16 = (_Float16*)alloc(A16);
  float*    h32   = (float*)   alloc(A32);
  _Float16* h16   = (_Float16*)alloc(A16);
  float*    sum32 = (float*)   alloc(A32);
  float*    xz32  = (float*)   alloc(2 * A32);
  float*    xc32  = (float*)   alloc(A32);
  _Float16* xc16  = (_Float16*)alloc(A16);
  float*    dbl32 = (float*)   alloc((size_t)kTT * 64 * sizeof(float));
  _Float16* dbl16 = (_Float16*)alloc((size_t)kTT * 64 * sizeof(_Float16));
  float*    dt32  = (float*)   alloc(A32);
  _Float16* y16   = (_Float16*)alloc(A16);
  float*    hn32  = (float*)   alloc(A32);
  _Float16* hn16  = (_Float16*)alloc(A16);
  _Float16* u16   = (_Float16*)alloc(A16);
  float*    dec32 = (float*)   alloc((size_t)kTT * kPred * sizeof(float));
  float*    meanb = (float*)   alloc((size_t)kB * kTok * sizeof(float));
  float*    stdb  = (float*)   alloc((size_t)kB * kTok * sizeof(float));

  _Float16* emb_w16  = (_Float16*)alloc((size_t)kD * kSeq * 2);
  _Float16* win16    = (_Float16*)alloc((size_t)4 * 2 * kD * kD * 2);
  _Float16* wx16     = (_Float16*)alloc((size_t)4 * (kR + 2 * kS) * kD * 2);
  _Float16* wdt16    = (_Float16*)alloc((size_t)4 * kD * kR * 2);
  _Float16* wout16   = (_Float16*)alloc((size_t)4 * kD * kD * 2);
  _Float16* ff1w16   = (_Float16*)alloc((size_t)2 * kD * kD * 2);
  _Float16* ff2w16   = (_Float16*)alloc((size_t)2 * kD * kD * 2);
  _Float16* projw16  = (_Float16*)alloc((size_t)kPred * kD * 2);

  auto cast = [&](const float* src, _Float16* dst, int n) {
    cast_f16_kernel<<<(n + 1023) / 1024, 256, 0, stream>>>(src, dst, n);
  };
  cast(emb_w,  emb_w16,  kD * kSeq);
  cast(m_win,  win16,    4 * 2 * kD * kD);
  cast(m_wx,   wx16,     4 * (kR + 2 * kS) * kD);
  cast(m_wdt,  wdt16,    4 * kD * kR);
  cast(m_wout, wout16,   4 * kD * kD);
  cast(ff1_w,  ff1w16,   2 * kD * kD);
  cast(ff2_w,  ff2w16,   2 * kD * kD);
  cast(proj_w, projw16,  kPred * kD);

  // ---- RevIN + token build ----
  norm_tokens_kernel<<<kB * kTok, 256, 0, stream>>>(x_enc, x_mark, tok16,
                                                    meanb, stdb);

  // ---- embedding: h = tok @ emb_w^T + emb_b ----
  launch_gemm(stream, tok16, kSeq, emb_w16, kSeq, emb_b,
              h32, kD, h16, kD, kTT, kD, kSeq, 0, 0);

  const int nElem = kTT * kD;
  for (int l = 0; l < 2; ++l) {
    // residual accumulator starts at h; fwd/rev mamba GEMMs add into it
    (void)hipMemcpyAsync(sum32, h32, A32, hipMemcpyDeviceToDevice, stream);

    for (int dir = 0; dir < 2; ++dir) {
      const int sl = l * 2 + dir;
      // xz = h @ win^T   (M=8384, N=1024, K=512)
      launch_gemm(stream, h16, kD, win16 + (size_t)sl * 2 * kD * kD, kD,
                  nullptr, xz32, 2 * kD, nullptr, 0, kTT, 2 * kD, kD, 0, 0);
      // conv(2-tap causal, direction-aware) + SiLU
      conv_silu_kernel<<<(nElem + 255) / 256, 256, 0, stream>>>(
          xz32, m_convw + (size_t)sl * kD * 2, m_convb + (size_t)sl * kD,
          xc32, xc16, dir);
      // dbl = xc @ wx^T  (N=64: dt_rank + B + C)
      launch_gemm(stream, xc16, kD, wx16 + (size_t)sl * (kR + 2 * kS) * kD, kD,
                  nullptr, dbl32, kR + 2 * kS, dbl16, kR + 2 * kS,
                  kTT, kR + 2 * kS, kD, 0, 0);
      // dt = softplus(dbl[:, :32] @ wdt^T + bdt)
      launch_gemm(stream, dbl16, kR + 2 * kS, wdt16 + (size_t)sl * kD * kR, kR,
                  m_bdt + (size_t)sl * kD, dt32, kD, nullptr, 0,
                  kTT, kD, kR, 2, 0);
      // selective scan + gating -> y16
      mamba_scan_kernel<<<kB, 512, 0, stream>>>(
          dt32, xc32, xz32, dbl32, m_alog + (size_t)sl * kD * kS,
          m_dparam + (size_t)sl * kD, y16, dir);
      // sum += y @ wout^T
      launch_gemm(stream, y16, kD, wout16 + (size_t)sl * kD * kD, kD,
                  nullptr, sum32, kD, nullptr, 0, kTT, kD, kD, 0, 1);
    }

    // hn = LN1(h + f + r)
    layernorm_kernel<<<kTT, 256, 0, stream>>>(
        sum32, ln1_g + l * kD, ln1_b + l * kD, hn32, hn16);
    // u = relu(hn @ ff1^T + b1)   (f16 only, feeds ff2)
    launch_gemm(stream, hn16, kD, ff1w16 + (size_t)l * kD * kD, kD,
                ff1_b + l * kD, nullptr, 0, u16, kD, kTT, kD, kD, 1, 0);
    // hn += u @ ff2^T + b2
    launch_gemm(stream, u16, kD, ff2w16 + (size_t)l * kD * kD, kD,
                ff2_b + l * kD, hn32, kD, nullptr, 0, kTT, kD, kD, 0, 1);
    // h = LN2(hn)
    layernorm_kernel<<<kTT, 256, 0, stream>>>(
        hn32, ln2_g + l * kD, ln2_b + l * kD, h32, h16);
  }

  // final LN -> f16 (reuse hn16), projection, transpose + denorm
  layernorm_kernel<<<kTT, 256, 0, stream>>>(h32, lnf_g, lnf_b, nullptr, hn16);
  launch_gemm(stream, hn16, kD, projw16, kD, proj_b,
              dec32, kPred, nullptr, 0, kTT, kPred, kD, 0, 0);
  final_denorm_kernel<<<(kB * kPred * kNvar + 255) / 256, 256, 0, stream>>>(
      dec32, meanb, stdb, out);
}
